// FastSpeech_15857019257480
// MI455X (gfx1250) — matrix operand order repaired and checked by reading, not compiled, not run on hardware
//
#include <hip/hip_runtime.h>
#include <hip/hip_bf16.h>
#include <math.h>

// ---------------------------------------------------------------------------
// FastSpeech forward for gfx1250 (MI455X, wave32, WMMA).
// All dense GEMMs (QKV, Wo, conv1/conv2 via shifted passes, mel head) use
// v_wmma_f32_16x16x32_f16. Attention's softmax-over-axis-0 is fused into a
// streaming kernel (it is local per (s1,s2) across the 16 batch-head slices).
// NOTE: gemm_kernel requires K % 32 == 0 (true for all launches: 384, 1536).
// ---------------------------------------------------------------------------

typedef _Float16 v16h __attribute__((ext_vector_type(16)));
typedef float    v8f  __attribute__((ext_vector_type(8)));

#define BDIM 384   // model dim
#define NB   8     // batch
#define LSEQ 256   // encoder seq len
#define DSH  192   // per-head dim
#define CDIM 1536  // conv hidden
#define NMEL 80

// ---------------------------------------------------------------------------
// positional encoding value: pe[pos][d]  (reference indexes pe by BATCH index)
// ---------------------------------------------------------------------------
__device__ __forceinline__ float pe_val(int pos, int d) {
    float i2 = (float)(d & ~1);
    float f  = __expf(-i2 * (9.210340371976184f / (float)BDIM)); // ln(10000)
    float a  = (float)pos * f;
    return (d & 1) ? __cosf(a) : __sinf(a);
}

// ---------------------------------------------------------------------------
// Embedding + batch-indexed PE
// ---------------------------------------------------------------------------
__global__ __launch_bounds__(256) void embed_kernel(const int* __restrict__ tokens,
                                                    const float* __restrict__ emb,
                                                    float* __restrict__ x) {
    int idx = blockIdx.x * 256 + threadIdx.x;
    if (idx >= NB * LSEQ * BDIM) return;
    int d  = idx % BDIM;
    int bl = idx / BDIM;
    int b  = bl / LSEQ;
    int tok = tokens[bl];
    x[idx] = emb[tok * BDIM + d] + pe_val(b, d);
}

// ---------------------------------------------------------------------------
// Tiled WMMA GEMM:  C[bz] (+)= act( A[bz](MxK) * W(KxN) + bias )
//   A rows optionally shifted by `shift` (zero pad OOB) -> exact SAME conv.
//   W addressed with generic strides (wsk over K, wsn over N) -> conv filters.
//   Block tile 64x64, 8 waves; each wave: one A frag, two 16x16 WMMA tiles.
//   Requires K % 32 == 0. Branchless bounds handling (clamp + select-zero).
// ---------------------------------------------------------------------------
__global__ __launch_bounds__(256) void gemm_kernel(
    const float* __restrict__ A, long a_batch, int lda, int M, int K,
    const float* __restrict__ W, long wsk, long wsn, int N,
    const float* __restrict__ bias,
    float* __restrict__ C, long c_batch, int ldc,
    int shift, int acc, int relu)
{
    __shared__ __align__(16) _Float16 Ash[64 * 32];
    __shared__ __align__(16) _Float16 Wsh[64 * 32];   // transposed [n][k]

    const int t    = threadIdx.x;
    const int m0   = blockIdx.y * 64;
    const int n0   = blockIdx.x * 64;
    const float* Ab = A + a_batch * (long)blockIdx.z;
    float*       Cb = C + c_batch * (long)blockIdx.z;

    const int wave = t >> 5, lane = t & 31;
    const int rw = wave & 3, cw = wave >> 2;   // 4 row tiles x 2 col halves
    const unsigned int* Au = (const unsigned int*)Ash;
    const unsigned int* Wu = (const unsigned int*)Wsh;

    v8f cacc0 = {0.f, 0.f, 0.f, 0.f, 0.f, 0.f, 0.f, 0.f};
    v8f cacc1 = {0.f, 0.f, 0.f, 0.f, 0.f, 0.f, 0.f, 0.f};

    // A staging: thread -> (row 0..63, 8 consecutive k)
    const int arow = t >> 2;
    const int ak0  = (t & 3) * 8;
    const int ar   = m0 + arow + shift;
    const bool rok = (ar >= 0) && (ar < M);
    const int arc  = rok ? ar : 0;
    // W staging: thread -> (n 0..63, 8 consecutive k); adjacent lanes adjacent n
    const int wn   = t & 63;
    const int wk0  = (t >> 6) * 8;
    const bool nok = (n0 + wn) < N;
    const int wnc  = nok ? (n0 + wn) : 0;

    // fragment index precompute (all 16B aligned dword runs)
    const int lan16 = lane & 15;
    const int gl    = lane >> 4;
    const int mrow  = rw * 16 + lan16;
    const int nl0   = cw * 32 + lan16;       // B frag 0 local col
    const int nl1   = nl0 + 16;              // B frag 1 local col
    const uint4* Aq  = (const uint4*)(Au + mrow * 16 + gl * 4);   // +0, +2
    const uint4* Bq0 = (const uint4*)(Wu + nl0 * 16 + gl * 4);    // +0, +2
    const uint4* Bq1 = (const uint4*)(Wu + nl1 * 16 + gl * 4);    // +0, +2

    for (int k0 = 0; k0 < K; k0 += 32) {
        __syncthreads();
        // ---- stage A tile (fp32 -> f16), row-major [64][32]
        {
            const float* Ar = Ab + (long)arc * lda + k0 + ak0;
            float4 a0 = *(const float4*)Ar;
            float4 a1 = *(const float4*)(Ar + 4);
            if (k0 + 32 < K) __builtin_prefetch(Ar + 32, 0, 3);  // global_prefetch_b8
            float s = rok ? 1.f : 0.f;
            union { _Float16 h[8]; uint4 u4; } pk;
            pk.h[0] = (_Float16)(a0.x * s); pk.h[1] = (_Float16)(a0.y * s);
            pk.h[2] = (_Float16)(a0.z * s); pk.h[3] = (_Float16)(a0.w * s);
            pk.h[4] = (_Float16)(a1.x * s); pk.h[5] = (_Float16)(a1.y * s);
            pk.h[6] = (_Float16)(a1.z * s); pk.h[7] = (_Float16)(a1.w * s);
            *(uint4*)(&Ash[arow * 32 + ak0]) = pk.u4;
        }
        // ---- stage W tile transposed [n][k] (fp32 -> f16)
        {
            const float* Wp = W + (long)(k0 + wk0) * wsk + (long)wnc * wsn;
            if (k0 + 32 < K) __builtin_prefetch(Wp + 32 * wsk, 0, 3);
            float s = nok ? 1.f : 0.f;
            union { _Float16 h[8]; uint4 u4; } pk;
            #pragma unroll
            for (int i = 0; i < 8; ++i)
                pk.h[i] = (_Float16)(Wp[(long)i * wsk] * s);
            *(uint4*)(&Wsh[wn * 32 + wk0]) = pk.u4;
        }
        __syncthreads();

        // ---- fragments (ISA 16-bit 16x32 A / 32x16 B layouts), b128 LDS loads
        union { v16h v; uint4 q[2]; } af, bf0, bf1;
        af.q[0]  = Aq[0];  af.q[1]  = Aq[2];   // A: dwords {base..+3, base+8..+11}
        bf0.q[0] = Bq0[0]; bf0.q[1] = Bq0[2];  // B: dwords {base..+3, base+8..+11}
        bf1.q[0] = Bq1[0]; bf1.q[1] = Bq1[2];
        cacc0 = __builtin_amdgcn_wmma_f32_16x16x32_f16(
            false, af.v, false, bf0.v, (short)0, cacc0, false, false);
        cacc1 = __builtin_amdgcn_wmma_f32_16x16x32_f16(
            false, af.v, false, bf1.v, (short)0, cacc1, false, false);
    }

    // ---- epilogue: C lane layout (N = lane%16, M = v + 8*(lane/16))
    #pragma unroll
    for (int half = 0; half < 2; ++half) {
        int col = n0 + (half ? nl1 : nl0);
        if (col < N) {
            #pragma unroll
            for (int vv = 0; vv < 8; ++vv) {
                int row = m0 + rw * 16 + vv + 8 * gl;
                if (row < M) {
                    float r = half ? cacc1[vv] : cacc0[vv];
                    long ci = (long)row * ldc + col;
                    if (acc)       r += Cb[ci];
                    else if (bias) r += bias[col];
                    if (relu)      r = fmaxf(r, 0.f);
                    Cb[ci] = r;
                }
            }
        }
    }
}

// ---------------------------------------------------------------------------
// Fused attention with softmax over axis 0 (across the 16 (b,h) slices).
// q,k,v layout (b, s, 384) with head h in cols [h*192, (h+1)*192).
// One block = 4 consecutive s1 rows for ALL 16 slices; streams over s2.
// ---------------------------------------------------------------------------
__global__ __launch_bounds__(256) void attn_kernel(const float* __restrict__ q,
                                                   const float* __restrict__ k,
                                                   const float* __restrict__ v,
                                                   float* __restrict__ z, int S) {
    __shared__ __align__(16) float qs[16][4][DSH];   // 48 KB
    __shared__ float sc[16][16][4];                  // raw scores [g][s2l][s1l]
    __shared__ float wsm[16][16][4];                 // softmax weights

    const int s1base = blockIdx.x * 4;
    const int t = threadIdx.x;

    for (int i = t; i < 16 * 4 * DSH; i += 256) {
        int g = i / (4 * DSH); int r = i % (4 * DSH);
        int s1l = r / DSH;     int d = r % DSH;
        int b = g >> 1, h = g & 1, s1 = s1base + s1l;
        qs[g][s1l][d] = (s1 < S) ? q[((long)b * S + s1) * BDIM + h * DSH + d] : 0.f;
    }
    __syncthreads();

    const int g = t >> 4;          // slice 0..15
    const int lane16 = t & 15;     // s2l (scores) / d-chunk (accum)
    const int b = g >> 1, h = g & 1;
    const float scale = 0.07216878364870323f;   // 1/sqrt(192)

    float acc[4][12];
    #pragma unroll
    for (int i = 0; i < 4; ++i)
        #pragma unroll
        for (int j = 0; j < 12; ++j) acc[i][j] = 0.f;

    for (int s2t = 0; s2t < S; s2t += 16) {
        // ---- scores: each thread one (g, s2), 4 s1 dots of length 192
        int s2 = s2t + lane16;
        float d0 = 0.f, d1 = 0.f, d2 = 0.f, d3 = 0.f;
        if (s2 < S) {
            const float4* kp = (const float4*)(k + ((long)b * S + s2) * BDIM + h * DSH);
            const float4* q0 = (const float4*)qs[g][0];
            const float4* q1 = (const float4*)qs[g][1];
            const float4* q2 = (const float4*)qs[g][2];
            const float4* q3 = (const float4*)qs[g][3];
            for (int d4 = 0; d4 < DSH / 4; ++d4) {
                float4 kv = kp[d4];
                float4 a0 = q0[d4], a1 = q1[d4], a2 = q2[d4], a3 = q3[d4];
                d0 += a0.x * kv.x + a0.y * kv.y + a0.z * kv.z + a0.w * kv.w;
                d1 += a1.x * kv.x + a1.y * kv.y + a1.z * kv.z + a1.w * kv.w;
                d2 += a2.x * kv.x + a2.y * kv.y + a2.z * kv.z + a2.w * kv.w;
                d3 += a3.x * kv.x + a3.y * kv.y + a3.z * kv.z + a3.w * kv.w;
            }
        }
        sc[g][lane16][0] = d0 * scale;
        sc[g][lane16][1] = d1 * scale;
        sc[g][lane16][2] = d2 * scale;
        sc[g][lane16][3] = d3 * scale;
        __syncthreads();

        // ---- softmax across g (16 values) per (s2,s1)
        #pragma unroll
        for (int s1l = 0; s1l < 4; ++s1l) {
            float mx = -1e30f;
            for (int gg = 0; gg < 16; ++gg) mx = fmaxf(mx, sc[gg][lane16][s1l]);
            float sum = 0.f;
            for (int gg = 0; gg < 16; ++gg) sum += __expf(sc[gg][lane16][s1l] - mx);
            wsm[g][lane16][s1l] = __expf(sc[g][lane16][s1l] - mx) / sum;
        }
        __syncthreads();

        // ---- accumulate z: thread owns (g, 12-wide d chunk), all 4 s1
        for (int s2l = 0; s2l < 16; ++s2l) {
            int ss = s2t + s2l;
            if (ss >= S) break;
            const float4* vp = (const float4*)(v + ((long)b * S + ss) * BDIM
                                               + h * DSH + lane16 * 12);
            float4 v0 = vp[0], v1 = vp[1], v2 = vp[2];
            float vv[12] = {v0.x, v0.y, v0.z, v0.w, v1.x, v1.y,
                            v1.z, v1.w, v2.x, v2.y, v2.z, v2.w};
            #pragma unroll
            for (int s1l = 0; s1l < 4; ++s1l) {
                float wv = wsm[g][s2l][s1l];
                #pragma unroll
                for (int j = 0; j < 12; ++j) acc[s1l][j] += wv * vv[j];
            }
        }
        __syncthreads();
    }

    #pragma unroll
    for (int s1l = 0; s1l < 4; ++s1l) {
        int s1 = s1base + s1l;
        if (s1 < S) {
            float* zp = z + ((long)b * S + s1) * BDIM + h * DSH + lane16 * 12;
            #pragma unroll
            for (int j = 0; j < 12; ++j) zp[j] = acc[s1l][j];
        }
    }
}

// ---------------------------------------------------------------------------
// x = LayerNorm(x + r) * gamma + beta   (in-place on x); one block per row.
// ---------------------------------------------------------------------------
__global__ __launch_bounds__(128) void add_ln_kernel(float* __restrict__ x,
                                                     const float* __restrict__ r,
                                                     const float* __restrict__ gamma,
                                                     const float* __restrict__ beta) {
    __shared__ float sbuf[128], s2buf[128];
    long row = blockIdx.x;
    float* xp = x + row * BDIM;
    const float* rp = r + row * BDIM;
    int tid = threadIdx.x;

    float vals[3], s = 0.f, s2 = 0.f;
    #pragma unroll
    for (int i = 0; i < 3; ++i) {
        int d = tid + i * 128;
        float vv = xp[d] + rp[d];
        vals[i] = vv; s += vv; s2 += vv * vv;
    }
    sbuf[tid] = s; s2buf[tid] = s2;
    __syncthreads();
    for (int off = 64; off > 0; off >>= 1) {
        if (tid < off) { sbuf[tid] += sbuf[tid + off]; s2buf[tid] += s2buf[tid + off]; }
        __syncthreads();
    }
    float mean = sbuf[0] * (1.f / BDIM);
    float var  = s2buf[0] * (1.f / BDIM) - mean * mean;
    float inv  = rsqrtf(var + 1e-5f);
    #pragma unroll
    for (int i = 0; i < 3; ++i) {
        int d = tid + i * 128;
        xp[d] = (vals[i] - mean) * inv * gamma[d] + beta[d];
    }
}

// ---------------------------------------------------------------------------
// Length regulation + decoder PE:  xd[b,t,:] = xe[b,j(t),:] (or 0) + pe[b,:]
// ---------------------------------------------------------------------------
__global__ __launch_bounds__(384) void regulate_kernel(const float* __restrict__ xe,
                                                       const int* __restrict__ dur,
                                                       float* __restrict__ xd, int T) {
    __shared__ int jj;
    int t = blockIdx.x, b = blockIdx.y;
    if (threadIdx.x == 0) {
        int cum = 0, j = -1;
        for (int i = 0; i < LSEQ; ++i) {
            int nx = cum + dur[b * LSEQ + i];
            if (t >= cum && t < nx) { j = i; break; }
            cum = nx;
        }
        jj = j;
    }
    __syncthreads();
    int d = threadIdx.x;
    int j = jj;
    float vv = (j >= 0) ? xe[((long)b * LSEQ + j) * BDIM + d] : 0.f;
    xd[((long)b * T + t) * BDIM + d] = vv + pe_val(b, d);
}

// ---------------------------------------------------------------------------
// Host-side orchestration
// ---------------------------------------------------------------------------
static void launch_gemm(hipStream_t st, const float* A, long ab, int lda, int M, int K,
                        const float* W, long wsk, long wsn, int N, const float* bias,
                        float* C, long cb, int ldc, int shift, int acc, int relu) {
    dim3 grid((N + 63) / 64, (M + 63) / 64, NB);
    gemm_kernel<<<grid, 256, 0, st>>>(A, ab, lda, M, K, W, wsk, wsn, N, bias,
                                      C, cb, ldc, shift, acc, relu);
}

static void run_fft_block(hipStream_t st, float* x, float* q, float* k, float* v,
                          float* z, float* z2, float* h, float* cout, int S,
                          const float* Wqkv, const float* bqkv,
                          const float* Wo, const float* bo, const float* ln1,
                          const float* c1w, const float* c1b,
                          const float* c2w, const float* c2b, const float* ln2) {
    long xs = (long)S * BDIM;
    long hs = (long)S * CDIM;
    float* qkv[3] = {q, k, v};
    for (int t = 0; t < 3; ++t)
        launch_gemm(st, x, xs, BDIM, S, BDIM,
                    Wqkv + (long)t * BDIM * BDIM, BDIM, 1, BDIM,
                    bqkv + t * BDIM, qkv[t], xs, BDIM, 0, 0, 1);
    attn_kernel<<<dim3((S + 3) / 4), 256, 0, st>>>(q, k, v, z, S);
    launch_gemm(st, z, xs, BDIM, S, BDIM, Wo, BDIM, 1, BDIM, bo, z2, xs, BDIM, 0, 0, 1);
    add_ln_kernel<<<NB * S, 128, 0, st>>>(x, z2, ln1, ln1 + BDIM);
    for (int kk = 0; kk < 3; ++kk)   // conv1: 384 -> 1536, SAME, K=3
        launch_gemm(st, x, xs, BDIM, S, BDIM,
                    c1w + kk, 3, (long)BDIM * 3, CDIM, c1b,
                    h, hs, CDIM, kk - 1, kk > 0, kk == 2);
    for (int kk = 0; kk < 3; ++kk)   // conv2: 1536 -> 384, SAME, K=3
        launch_gemm(st, h, hs, CDIM, S, CDIM,
                    c2w + kk, 3, (long)CDIM * 3, BDIM, c2b,
                    cout, xs, BDIM, kk - 1, kk > 0, kk == 2);
    add_ln_kernel<<<NB * S, 128, 0, st>>>(x, cout, ln2, ln2 + BDIM);
}

extern "C" void kernel_launch(void* const* d_in, const int* in_sizes, int n_in,
                              void* d_out, int out_size, void* d_ws, size_t ws_size,
                              hipStream_t stream) {
    (void)in_sizes; (void)n_in; (void)ws_size;
    const int*   tokens = (const int*)d_in[0];
    const int*   target = (const int*)d_in[1];
    const float* emb    = (const float*)d_in[2];
    const float* eW[10], * dW[10];
    for (int i = 0; i < 10; ++i) { eW[i] = (const float*)d_in[3 + i]; dW[i] = (const float*)d_in[13 + i]; }
    const float* head_w = (const float*)d_in[23];
    const float* head_b = (const float*)d_in[24];

    const int T = out_size / (NB * NMEL);           // expand_max_len
    const int Smax = (T > LSEQ) ? T : LSEQ;
    const long P = (long)NB * Smax * BDIM;          // one slot, floats
    float* ws = (float*)d_ws;

    // ---------------- encoder (S = 256), x in slot0
    embed_kernel<<<(NB * LSEQ * BDIM + 255) / 256, 256, 0, stream>>>(tokens, emb, ws);
    for (int l = 0; l < 4; ++l)
        run_fft_block(stream,
                      /*x*/ ws, /*q*/ ws + 2 * P, /*k*/ ws + 3 * P, /*v*/ ws + 4 * P,
                      /*z*/ ws + P, /*z2*/ ws + 2 * P, /*h*/ ws + 2 * P, /*cout*/ ws + P,
                      LSEQ,
                      eW[0] + (long)l * 3 * BDIM * BDIM, eW[1] + (long)l * 3 * BDIM,
                      eW[2] + (long)l * BDIM * BDIM,     eW[3] + (long)l * BDIM,
                      eW[4] + (long)l * 2 * BDIM,
                      eW[5] + (long)l * CDIM * BDIM * 3, eW[6] + (long)l * CDIM,
                      eW[7] + (long)l * BDIM * CDIM * 3, eW[8] + (long)l * BDIM,
                      eW[9] + (long)l * 2 * BDIM);

    // ---------------- length regulate + decoder PE, x -> slot1
    regulate_kernel<<<dim3(T, NB), 384, 0, stream>>>(ws, target, ws + P, T);

    // ---------------- decoder (S = T), x in slot1
    for (int l = 0; l < 4; ++l)
        run_fft_block(stream,
                      /*x*/ ws + P, /*q*/ ws, /*k*/ ws + 2 * P, /*v*/ ws + 3 * P,
                      /*z*/ ws + 4 * P, /*z2*/ ws, /*h*/ ws + 2 * P, /*cout*/ ws,
                      T,
                      dW[0] + (long)l * 3 * BDIM * BDIM, dW[1] + (long)l * 3 * BDIM,
                      dW[2] + (long)l * BDIM * BDIM,     dW[3] + (long)l * BDIM,
                      dW[4] + (long)l * 2 * BDIM,
                      dW[5] + (long)l * CDIM * BDIM * 3, dW[6] + (long)l * CDIM,
                      dW[7] + (long)l * BDIM * CDIM * 3, dW[8] + (long)l * BDIM,
                      dW[9] + (long)l * 2 * BDIM);

    // ---------------- mel head: (B,T,384) @ (384,80) + b
    launch_gemm(stream, ws + P, (long)T * BDIM, BDIM, T, BDIM,
                head_w, NMEL, 1, NMEL, head_b,
                (float*)d_out, (long)T * NMEL, NMEL, 0, 0, 0);
}
